// GatedDeltaNet_82514911691033
// MI455X (gfx1250) — matrix operand order repaired and checked by reading, not compiled, not run on hardware
//
#include <hip/hip_runtime.h>
#include <hip/hip_bf16.h>
#include <math.h>

// ---------------------------------------------------------------------------
// GatedDeltaNet forward for MI455X (gfx1250, wave32, WMMA).
// B=2, S=2048, HID=2048, H=16, D=128, K(window/conv)=4.
// Heavy GEMMs run as bf16 WMMA (v_wmma_f32_16x16x32_bf16) with f32 accum,
// B tiles staged through LDS (shared by all 8 waves of the block).
// ---------------------------------------------------------------------------

#define BB   2
#define SS   2048
#define HID  2048
#define NH   16
#define HD   128
#define KW   4
#define MROWS (BB * SS)          // 4096
#define CCH  (NH * HD)           // 2048

typedef __attribute__((ext_vector_type(16))) __bf16 v16bf;
typedef __attribute__((ext_vector_type(8)))  __bf16 v8bf;
typedef __attribute__((ext_vector_type(8)))  float  v8f;

union bffrag { v16bf v; v8bf h[2]; };

// Fast sigmoid/silu: v_exp_f32 + v_rcp_f32 (bf16-pipeline accuracy is ample).
__device__ __forceinline__ float fsigmoid(float x) {
  return __builtin_amdgcn_rcpf(1.0f + __expf(-x));
}
__device__ __forceinline__ float fsilu(float x) {
  return x * __builtin_amdgcn_rcpf(1.0f + __expf(-x));
}

// --------------------------- fp32 -> bf16 convert ---------------------------
__global__ __launch_bounds__(256) void k_cvt_bf16(const float* __restrict__ in,
                                                  __bf16* __restrict__ out, long n) {
  long i = (long)blockIdx.x * blockDim.x + threadIdx.x;
  if (i < n) out[i] = (__bf16)in[i];
}

// W[k*N + n] (fp32, row-major [K,N]) -> Wt[n*K + k] (bf16, K-major per output col)
__global__ __launch_bounds__(256) void k_cvt_transpose_bf16(const float* __restrict__ W,
                                                            __bf16* __restrict__ Wt,
                                                            int K, int N) {
  long i = (long)blockIdx.x * blockDim.x + threadIdx.x;
  long total = (long)K * N;
  if (i >= total) return;
  int n = (int)(i % N);
  int k = (int)(i / N);
  Wt[(size_t)n * K + k] = (__bf16)W[(size_t)k * N + n];
}

// ------------------------------ WMMA GEMM ----------------------------------
// C[M,N] = A[M,Kd] * Bt[N,Kd]^T ; optional fused silu epilogue.
// Block = 256 threads = 8 waves; wave w owns rows m0 = by*128 + w*16,
// cols n0 = bx*(NTILES*16). B tile staged in LDS once per k-step, shared by
// all 8 waves; A fragments loaded straight from global (two b128 per lane).
// Requires: M % 128 == 0, N % (NTILES*16) == 0, Kd % 32 == 0.
template <int NTILES, bool DOSILU>
__global__ __launch_bounds__(256) void k_gemm_bf16_wmma(const __bf16* __restrict__ A,
                                                        const __bf16* __restrict__ Bt,
                                                        float* __restrict__ C,
                                                        int M, int N, int Kd) {
  __shared__ __bf16 bsh[NTILES * 16 * 32];     // [col][k] for current 32-wide k-step

  const int lane = threadIdx.x & 31;
  const int wave = threadIdx.x >> 5;
  const int m0 = blockIdx.y * 128 + wave * 16;
  const int n0 = blockIdx.x * (NTILES * 16);
  const int l16 = lane & 15;
  const int hi  = lane >> 4;

  const __bf16* aRow = A + (size_t)(m0 + l16) * Kd;

  // B staging assignment: thread -> (col, 8-elem segment)
  const int scol = threadIdx.x >> 2;           // 0 .. 63
  const int sseg = threadIdx.x & 3;            // 0 .. 3
  const __bf16* bSrc = Bt + (size_t)(n0 + scol) * Kd + sseg * 8;
  __bf16* bDst = &bsh[scol * 32 + sseg * 8];

  v8f acc[NTILES] = {};

  #pragma unroll 1
  for (int k0 = 0; k0 < Kd; k0 += 32) {
    // ---- cooperative B tile load: global -> LDS (4KB for NTILES=4) ----
    // For NTILES==4 every thread stages (guard folds away at compile time).
    if (NTILES == 4 || threadIdx.x < NTILES * 64)
      *(v8bf*)bDst = *(const v8bf*)(bSrc + k0);
    __syncthreads();

    // ---- A fragment (16x32 bf16): two contiguous 16B loads per lane ----
    // Lane (l16,hi): VGPR0-3 hold K = hi*8+0..7, VGPR4-7 hold K = 16+hi*8+0..7
    bffrag af;
    af.h[0] = *(const v8bf*)(aRow + k0 + hi * 8);
    af.h[1] = *(const v8bf*)(aRow + k0 + 16 + hi * 8);
    __builtin_prefetch((const void*)(aRow + k0 + 64), 0, 3);

    // ---- NTILES WMMAs, B fragments from LDS (two ds_load_b128 each) ----
    #pragma unroll
    for (int t = 0; t < NTILES; ++t) {
      const __bf16* bp = &bsh[(t * 16 + l16) * 32 + hi * 8];
      bffrag bfg;
      bfg.h[0] = *(const v8bf*)(bp);
      bfg.h[1] = *(const v8bf*)(bp + 16);
      acc[t] = __builtin_amdgcn_wmma_f32_16x16x32_bf16(
          false, af.v, false, bfg.v, (short)0, acc[t], false, false);
    }
    __syncthreads();
  }

  // C/D layout: lane col = l16, VGPR r -> row m0 + hi*8 + r.
  #pragma unroll
  for (int t = 0; t < NTILES; ++t) {
    const int col = n0 + t * 16 + l16;
    #pragma unroll
    for (int r = 0; r < 8; ++r) {
      float vv = acc[t][r];
      if (DOSILU) vv = fsilu(vv);
      C[(size_t)(m0 + hi * 8 + r) * N + col] = vv;
    }
  }
}

// ---------------------- causal depthwise conv (K=4) ------------------------
// y[b,s,c] = sum_{lag=0..3} w[c, 3-lag] * x[b, s-lag, c]
__global__ __launch_bounds__(256) void k_dwconv(const float* __restrict__ in,
                                                const float* __restrict__ w,
                                                float* __restrict__ out,
                                                int S, int C, long total) {
  long idx = (long)blockIdx.x * blockDim.x + threadIdx.x;
  if (idx >= total) return;
  int c = (int)(idx % C);
  long sc = idx / C;
  int s = (int)(sc % S);
  float acc = 0.0f;
  #pragma unroll
  for (int lag = 0; lag < KW; ++lag) {
    if (s - lag >= 0) acc += w[c * KW + (KW - 1 - lag)] * in[idx - (long)lag * C];
  }
  out[idx] = acc;
}

// --------------------- RoPE + L2-normalize (q and k) -----------------------
// One wave32 per (tensor, b, s, h) row of 128 elems; in-place.
__global__ __launch_bounds__(256) void k_rope_l2norm(float* __restrict__ q,
                                                     float* __restrict__ k,
                                                     int BSH) {
  int wv = (int)((blockIdx.x * (long)blockDim.x + threadIdx.x) >> 5);
  int lane = threadIdx.x & 31;
  if (wv >= 2 * BSH) return;
  float* base = (wv < BSH) ? q : k;
  int r = (wv < BSH) ? wv : wv - BSH;
  int h  = r % NH;
  int bs = r / NH;
  int s  = bs % SS;
  float* p = base + (size_t)bs * CCH + h * HD;

  const float lg = logf(1.0e6f);
  float y[4];
  float ss = 0.0f;
  #pragma unroll
  for (int j = 0; j < 2; ++j) {
    int pd = lane + 32 * j;                       // half-dim index 0..63
    float inv = __expf(-(2.0f * pd / (float)HD) * lg);
    float ang = (float)s * inv;
    float cs = cosf(ang), sn = sinf(ang);
    float u1 = p[pd], u2 = p[pd + 64];
    float y1 = u1 * cs - u2 * sn;
    float y2 = u2 * cs + u1 * sn;
    y[2 * j] = y1; y[2 * j + 1] = y2;
    ss += y1 * y1 + y2 * y2;
  }
  #pragma unroll
  for (int off = 16; off > 0; off >>= 1) ss += __shfl_xor(ss, off, 32);
  float scl = 1.0f / fmaxf(sqrtf(ss), 1e-12f);
  #pragma unroll
  for (int j = 0; j < 2; ++j) {
    int pd = lane + 32 * j;
    p[pd]      = y[2 * j] * scl;
    p[pd + 64] = y[2 * j + 1] * scl;
  }
}

// ------------------- windowed gated delta attention (win=4) ----------------
// out[i] = sum_{lag} (q_i . k_{i-lag}) * scale * sigmoid(a_{i-lag} * b_i) * v_{i-lag}
__global__ __launch_bounds__(256) void k_window_attn(const float* __restrict__ q,
                                                     const float* __restrict__ k,
                                                     const float* __restrict__ v,
                                                     const float* __restrict__ a,
                                                     const float* __restrict__ b,
                                                     float* __restrict__ out,
                                                     int BSH) {
  int wv = (int)((blockIdx.x * (long)blockDim.x + threadIdx.x) >> 5);
  int lane = threadIdx.x & 31;
  if (wv >= BSH) return;
  int h  = wv % NH;
  int bs = wv / NH;
  int s  = bs % SS;
  size_t rowoff = (size_t)bs * CCH + h * HD;
  const float scale = 0.088388347648318447f;   // 1/sqrt(128)

  float qd[4];
  #pragma unroll
  for (int j = 0; j < 4; ++j) qd[j] = q[rowoff + lane + 32 * j];
  float bcur = b[(size_t)bs * NH + h];

  float acc[4] = {0.f, 0.f, 0.f, 0.f};
  #pragma unroll
  for (int lag = 0; lag < KW; ++lag) {
    if (s - lag < 0) break;                      // wave-uniform
    size_t loff = rowoff - (size_t)lag * CCH;
    float kd[4], vd[4];
    float dot = 0.0f;
    #pragma unroll
    for (int j = 0; j < 4; ++j) {
      kd[j] = k[loff + lane + 32 * j];
      vd[j] = v[loff + lane + 32 * j];
      dot += qd[j] * kd[j];
    }
    #pragma unroll
    for (int off = 16; off > 0; off >>= 1) dot += __shfl_xor(dot, off, 32);
    float alag = a[(size_t)(bs - lag) * NH + h];
    float wgt = dot * scale * fsigmoid(alag * bcur);
    #pragma unroll
    for (int j = 0; j < 4; ++j) acc[j] += wgt * vd[j];
  }
  #pragma unroll
  for (int j = 0; j < 4; ++j) out[rowoff + lane + 32 * j] = acc[j];
}

// -------------------- gated RMSNorm -> bf16 GEMM input ---------------------
__global__ __launch_bounds__(256) void k_rmsnorm_gate(const float* __restrict__ o,
                                                      const float* __restrict__ z,
                                                      const float* __restrict__ nw,
                                                      __bf16* __restrict__ out) {
  __shared__ float red[256];
  int row = blockIdx.x;
  const float* orow = o + (size_t)row * CCH;
  const float* zrow = z + (size_t)row * CCH;
  float s = 0.0f;
  for (int c = threadIdx.x; c < CCH; c += 256) { float u = orow[c]; s += u * u; }
  red[threadIdx.x] = s;
  __syncthreads();
  for (int off = 128; off > 0; off >>= 1) {
    if (threadIdx.x < off) red[threadIdx.x] += red[threadIdx.x + off];
    __syncthreads();
  }
  float rms = rsqrtf(red[0] / (float)CCH + 1e-6f);
  for (int c = threadIdx.x; c < CCH; c += 256) {
    float g = fsilu(zrow[c]);
    out[(size_t)row * CCH + c] = (__bf16)(orow[c] * rms * nw[c] * g);
  }
}

// ---------------------------------------------------------------------------
extern "C" void kernel_launch(void* const* d_in, const int* in_sizes, int n_in,
                              void* d_out, int out_size, void* d_ws, size_t ws_size,
                              hipStream_t stream) {
  (void)in_sizes; (void)n_in; (void)out_size; (void)ws_size;

  const float* x       = (const float*)d_in[0];   // [B,S,HID]
  const float* Wq      = (const float*)d_in[1];   // [HID, H*D]
  const float* Wk      = (const float*)d_in[2];
  const float* Wv      = (const float*)d_in[3];
  const float* Wz      = (const float*)d_in[4];
  const float* Wa      = (const float*)d_in[5];   // [HID, H]
  const float* Wb      = (const float*)d_in[6];
  const float* convq   = (const float*)d_in[7];   // [C, K]
  const float* convk   = (const float*)d_in[8];
  const float* convv   = (const float*)d_in[9];
  const float* norm_w  = (const float*)d_in[10];  // [C]
  const float* Wout    = (const float*)d_in[11];  // [C, HID]
  float* out           = (float*)d_out;           // [B,S,HID]

  // -------- workspace carve-up --------
  char* w = (char*)d_ws;
  size_t off = 0;
  auto alloc = [&](size_t bytes) { void* p = w + off; off += (bytes + 255) & ~(size_t)255; return p; };

  __bf16* xb  = (__bf16*)alloc((size_t)MROWS * HID * 2);   // x in bf16
  __bf16* wt  = (__bf16*)alloc((size_t)HID * CCH * 2);     // transposed bf16 weight (reused)
  float*  qf  = (float*) alloc((size_t)MROWS * CCH * 4);
  float*  kf  = (float*) alloc((size_t)MROWS * CCH * 4);
  float*  vf  = (float*) alloc((size_t)MROWS * CCH * 4);
  float*  zf  = (float*) alloc((size_t)MROWS * CCH * 4);
  float*  af  = (float*) alloc((size_t)MROWS * NH * 4);
  float*  bf  = (float*) alloc((size_t)MROWS * NH * 4);
  float*  qc  = (float*) alloc((size_t)MROWS * CCH * 4);
  float*  kc  = (float*) alloc((size_t)MROWS * CCH * 4);
  float*  vc  = (float*) alloc((size_t)MROWS * CCH * 4);
  float*  of  = (float*) alloc((size_t)MROWS * CCH * 4);
  __bf16* nb  = (__bf16*)alloc((size_t)MROWS * CCH * 2);

  const long nx = (long)MROWS * HID;
  const long nw_big = (long)HID * CCH;
  const long nw_sml = (long)HID * NH;

  dim3 blk(256);
  dim3 gGemmBig(CCH / 64, MROWS / 128);
  dim3 gGemmSml(1,        MROWS / 128);

  // 1) x -> bf16
  k_cvt_bf16<<<dim3((nx + 255) / 256), blk, 0, stream>>>(x, xb, nx);

  // 2) big projections: q,k,v (silu fused), z
  k_cvt_transpose_bf16<<<dim3((nw_big + 255) / 256), blk, 0, stream>>>(Wq, wt, HID, CCH);
  k_gemm_bf16_wmma<4, true ><<<gGemmBig, blk, 0, stream>>>(xb, wt, qf, MROWS, CCH, HID);
  k_cvt_transpose_bf16<<<dim3((nw_big + 255) / 256), blk, 0, stream>>>(Wk, wt, HID, CCH);
  k_gemm_bf16_wmma<4, true ><<<gGemmBig, blk, 0, stream>>>(xb, wt, kf, MROWS, CCH, HID);
  k_cvt_transpose_bf16<<<dim3((nw_big + 255) / 256), blk, 0, stream>>>(Wv, wt, HID, CCH);
  k_gemm_bf16_wmma<4, true ><<<gGemmBig, blk, 0, stream>>>(xb, wt, vf, MROWS, CCH, HID);
  k_cvt_transpose_bf16<<<dim3((nw_big + 255) / 256), blk, 0, stream>>>(Wz, wt, HID, CCH);
  k_gemm_bf16_wmma<4, false><<<gGemmBig, blk, 0, stream>>>(xb, wt, zf, MROWS, CCH, HID);

  // 3) small projections: a, b (N=16, no silu)
  k_cvt_transpose_bf16<<<dim3((nw_sml + 255) / 256), blk, 0, stream>>>(Wa, wt, HID, NH);
  k_gemm_bf16_wmma<1, false><<<gGemmSml, blk, 0, stream>>>(xb, wt, af, MROWS, NH, HID);
  k_cvt_transpose_bf16<<<dim3((nw_sml + 255) / 256), blk, 0, stream>>>(Wb, wt, HID, NH);
  k_gemm_bf16_wmma<1, false><<<gGemmSml, blk, 0, stream>>>(xb, wt, bf, MROWS, NH, HID);

  // 4) causal depthwise conv on q,k,v
  const long nconv = (long)MROWS * CCH;
  dim3 gConv((nconv + 255) / 256);
  k_dwconv<<<gConv, blk, 0, stream>>>(qf, convq, qc, SS, CCH, nconv);
  k_dwconv<<<gConv, blk, 0, stream>>>(kf, convk, kc, SS, CCH, nconv);
  k_dwconv<<<gConv, blk, 0, stream>>>(vf, convv, vc, SS, CCH, nconv);

  // 5) RoPE + L2 norm on q,k (in place); one wave per row, 2*B*S*H rows
  const int BSH = MROWS * NH;
  k_rope_l2norm<<<dim3((2 * BSH + 7) / 8), blk, 0, stream>>>(qc, kc, BSH);

  // 6) windowed gated delta attention; one wave per (b,s,h)
  k_window_attn<<<dim3((BSH + 7) / 8), blk, 0, stream>>>(qc, kc, vc, af, bf, of, BSH);

  // 7) gated RMSNorm -> bf16
  k_rmsnorm_gate<<<dim3(MROWS), blk, 0, stream>>>(of, zf, norm_w, nb);

  // 8) final GEMM: out = normed @ Wout
  k_cvt_transpose_bf16<<<dim3((nw_big + 255) / 256), blk, 0, stream>>>(Wout, wt, CCH, HID);
  k_gemm_bf16_wmma<4, false><<<gGemmBig, blk, 0, stream>>>(nb, wt, out, MROWS, HID, CCH);
}
